// MultiHeadAttention_48825188221343
// MI455X (gfx1250) — matrix operand order repaired and checked by reading, not compiled, not run on hardware
//
#include <hip/hip_runtime.h>
#include <hip/hip_bf16.h>
#include <cstdint>

// Problem constants (from reference)
#define BB   32
#define DQN  512
#define DKN  512
#define NKN  16
#define HN   8
#define TT   12
#define NNN  207
#define CC   64
#define TNC  (TT * NNN * CC)          // 158976, divisible by 256 (=621*256)
#define QCHUNKS 4
#define QT_PER_CHUNK 8                // 8 q-tiles of 16 => 128 q per chunk

typedef __attribute__((ext_vector_type(16))) __bf16 v16bf;
typedef __attribute__((ext_vector_type(8)))  float  v8f;
typedef __attribute__((ext_vector_type(4)))  float  v4f;
typedef __attribute__((ext_vector_type(4)))  unsigned int v4u;

union Frag32B {               // 32 bytes: one bf16 16-element WMMA fragment
    v16bf v;
    v4u   u[2];
};

__device__ __forceinline__ unsigned short f2bf_rne(float x) {
    unsigned int u = __float_as_uint(x);
    u += 0x7fffu + ((u >> 16) & 1u);  // round-to-nearest-even
    return (unsigned short)(u >> 16);
}

// ---------------------------------------------------------------------------
// Kernel 0: convert W -> bf16, build keys^T (b,k,d) -> bf16
// ---------------------------------------------------------------------------
__global__ __launch_bounds__(256)
void prep_kernel(const float* __restrict__ keys,   // [B, DK, NK]
                 const float* __restrict__ W,      // [H, DQ, DK]
                 unsigned short* __restrict__ Wb,  // [H, DQ, DK] bf16
                 unsigned short* __restrict__ At)  // [B, NK, DK] bf16 (keys^T)
{
    const int WN = HN * DQN * DKN;      // 2,097,152
    const int AN = BB * NKN * DKN;      //   262,144
    int stride = gridDim.x * blockDim.x;
    for (int i = blockIdx.x * blockDim.x + threadIdx.x; i < WN; i += stride)
        Wb[i] = f2bf_rne(W[i]);
    for (int i = blockIdx.x * blockDim.x + threadIdx.x; i < AN; i += stride) {
        int b = i / (NKN * DKN);
        int r = i - b * (NKN * DKN);
        int k = r / DKN;
        int d = r - k * DKN;
        At[i] = f2bf_rne(keys[((size_t)b * DKN + d) * NKN + k]);
    }
}

// ---------------------------------------------------------------------------
// Kernel 1: per (b,h,qchunk): WMMA GEMM K = keys^T * W^T, fused bias+relu+Q-dot
//   Schunk[b,h,qc,k] = sum_{q in chunk} Q[b,q] * relu(K[b,h,k,q] + bias[h,q])
// One wave per block. A tile: 16(k) x 32(d) bf16; B tile: 32(d) x 16(q) bf16.
// ---------------------------------------------------------------------------
__global__ __launch_bounds__(32)
void attn_gemm_kernel(const float* __restrict__ Q,            // [B, DQ]
                      const float* __restrict__ bias,         // [H, DQ]
                      const unsigned short* __restrict__ Wb,  // [H, DQ, DK] bf16
                      const unsigned short* __restrict__ At,  // [B, NK, DK] bf16
                      float* __restrict__ Schunk)             // [B, H, QCHUNKS, 16]
{
    int blk = blockIdx.x;                      // B*H*QCHUNKS
    int qc  = blk % QCHUNKS;
    int h   = (blk / QCHUNKS) % HN;
    int b   = blk / (QCHUNKS * HN);
    int lane = threadIdx.x;                    // wave32
    int n    = lane & 15;                      // q (col) within tile / k row for A
    int kh   = lane >> 4;                      // lane-half selector

    const int qcbase = qc * (QT_PER_CHUNK * 16);
    const unsigned short* Ab = At + (size_t)b * NKN * DKN + (size_t)n * DKN;
    const unsigned short* Wh = Wb + (size_t)h * DQN * DKN;

    v8f c[QT_PER_CHUNK] = {};                  // fp32 accumulator tiles

    for (int ds = 0; ds < DKN / 32; ++ds) {
        const int dbase = ds * 32;
        // A fragment (16-bit A 16x32 layout, ISA 7.12.2):
        //  lane l: row m=l&15; K = 8*kh + e (e<8), 16 + 8*kh + (e-8) (e>=8)
        Frag32B a;
        a.u[0] = *(const v4u*)(Ab + dbase + 8 * kh);
        a.u[1] = *(const v4u*)(Ab + dbase + 16 + 8 * kh);
#pragma unroll
        for (int t = 0; t < QT_PER_CHUNK; ++t) {
            int qg = qcbase + t * 16 + n;
            // B fragment: col n, K = 8*kh + e (e<8), 16 + 8*kh + (e-8) (e>=8) of W[h,qg,:]
            const unsigned short* bp = Wh + (size_t)qg * DKN + dbase + 8 * kh;
            Frag32B bf;
            bf.u[0] = *(const v4u*)bp;
            bf.u[1] = *(const v4u*)(bp + 16);
            c[t] = __builtin_amdgcn_wmma_f32_16x16x32_bf16(
                false, a.v, false, bf.v, (short)0, c[t], false, false);
        }
    }

    // Epilogue: C layout lane l: q-col = l&15, VGPR j -> k-row = j + 8*kh.
    float acc[8] = {0.f, 0.f, 0.f, 0.f, 0.f, 0.f, 0.f, 0.f};
#pragma unroll
    for (int t = 0; t < QT_PER_CHUNK; ++t) {
        int qg   = qcbase + t * 16 + n;
        float qv = Q[(size_t)b * DQN + qg];
        float bv = bias[(size_t)h * DQN + qg];
#pragma unroll
        for (int j = 0; j < 8; ++j)
            acc[j] += fmaxf(c[t][j] + bv, 0.f) * qv;
    }
    // reduce over the 16 lanes of each half (k rows are disjoint across halves)
#pragma unroll
    for (int m = 1; m < 16; m <<= 1)
#pragma unroll
        for (int j = 0; j < 8; ++j)
            acc[j] += __shfl_xor(acc[j], m, 32);

    if (n == 0) {  // lanes 0 and 16 hold the half-sums
        float* dst = Schunk + ((((size_t)b * HN + h) * QCHUNKS) + qc) * 16 + 8 * kh;
#pragma unroll
        for (int j = 0; j < 8; ++j) dst[j] = acc[j];
    }
}

// ---------------------------------------------------------------------------
// Kernel 2: att[b,k] = softmax_k( mean_h(S) / sqrt(512) )
// ---------------------------------------------------------------------------
__global__ __launch_bounds__(32)
void softmax_kernel(const float* __restrict__ Schunk,  // [B, H, QCHUNKS, 16]
                    float* __restrict__ att)           // [B, 16]
{
    int b = blockIdx.x;
    int lane = threadIdx.x;
    int k = lane & 15;
    float v = 0.f;
    if (lane < 16) {
        for (int h = 0; h < HN; ++h)
            for (int qc = 0; qc < QCHUNKS; ++qc)
                v += Schunk[((((size_t)b * HN + h) * QCHUNKS) + qc) * 16 + k];
        v *= (1.0f / HN) * 0.044194173824159216f;   // 1/sqrt(512)
    }
    float mx = v;
#pragma unroll
    for (int m = 1; m < 16; m <<= 1) mx = fmaxf(mx, __shfl_xor(mx, m, 16));
    float e = (lane < 16) ? __expf(v - mx) : 0.f;
    float s = e;
#pragma unroll
    for (int m = 1; m < 16; m <<= 1) s += __shfl_xor(s, m, 16);
    if (lane < 16) att[b * 16 + k] = e / s;
}

// ---------------------------------------------------------------------------
// Kernel 3: out[b,i] = sum_e att[b,e] * V[b,i,e]   (streaming, NT hints)
// ---------------------------------------------------------------------------
__global__ __launch_bounds__(256)
void out_kernel(const float* __restrict__ V,    // [B, TNC, 16]
                const float* __restrict__ att,  // [B, 16]
                float* __restrict__ out)        // [B, TNC]
{
    const int blocksPerB = TNC / 256;           // 621
    int b = blockIdx.x / blocksPerB;
    int i = (blockIdx.x % blocksPerB) * 256 + threadIdx.x;

    const v4f* ap = (const v4f*)(att + b * 16);    // uniform -> scalar loads
    v4f a0 = ap[0], a1 = ap[1], a2 = ap[2], a3 = ap[3];

    const v4f* vp = (const v4f*)(V + ((size_t)b * TNC + i) * 16);
    v4f v0 = __builtin_nontemporal_load(vp + 0);
    v4f v1 = __builtin_nontemporal_load(vp + 1);
    v4f v2 = __builtin_nontemporal_load(vp + 2);
    v4f v3 = __builtin_nontemporal_load(vp + 3);

    float r = v0.x * a0.x + v0.y * a0.y + v0.z * a0.z + v0.w * a0.w
            + v1.x * a1.x + v1.y * a1.y + v1.z * a1.z + v1.w * a1.w
            + v2.x * a2.x + v2.y * a2.y + v2.z * a2.z + v2.w * a2.w
            + v3.x * a3.x + v3.y * a3.y + v3.z * a3.z + v3.w * a3.w;

    __builtin_nontemporal_store(r, out + (size_t)b * TNC + i);
}

// ---------------------------------------------------------------------------
extern "C" void kernel_launch(void* const* d_in, const int* in_sizes, int n_in,
                              void* d_out, int out_size, void* d_ws, size_t ws_size,
                              hipStream_t stream) {
    const float* query = (const float*)d_in[0];   // [B, DQ]
    const float* keys  = (const float*)d_in[1];   // [B, DK, NK]
    const float* V     = (const float*)d_in[2];   // [B, T, N, C, NK]
    const float* W     = (const float*)d_in[3];   // [H, DQ, DK]
    const float* bias  = (const float*)d_in[4];   // [H, DQ]
    float* out = (float*)d_out;

    char* ws = (char*)d_ws;
    unsigned short* Wb = (unsigned short*)ws;                       // 4,194,304 B
    unsigned short* At = (unsigned short*)(ws + 4194304);           //   524,288 B
    float* Sc          = (float*)(ws + 4194304 + 524288);           //    65,536 B
    float* att         = (float*)(ws + 4194304 + 524288 + 32768);   //     2,048 B

    prep_kernel<<<1024, 256, 0, stream>>>(keys, W, Wb, At);
    attn_gemm_kernel<<<BB * HN * QCHUNKS, 32, 0, stream>>>(query, bias, Wb, At, Sc);
    softmax_kernel<<<BB, 32, 0, stream>>>(Sc, att);
    out_kernel<<<BB * (TNC / 256), 256, 0, stream>>>(V, att, out);
}
